// MultiheadLatentAttention_3272765080433
// MI455X (gfx1250) — matrix operand order repaired and checked by reading, not compiled, not run on hardware
//
#include <hip/hip_runtime.h>

#define BB  2
#define TT  2048
#define DDm 2048
#define HHn 16
#define HDn 128
#define DCn 512
#define DRn 64
#define MMr (BB * TT)   // 4096

// dynamic-LDS layout for the flash kernel (no static __shared__ => offset 0)
#define K_TILE_BYTES 12288           // 32 keys x 192 d x bf16, per buffer
#define V_TILE_OFF   24576           // after 2 K buffers
#define V_TILE_BYTES 8192            // 128 d x 32 keys x bf16, per buffer
#define P_OFF        40960           // after 2 V buffers
#define SMEM_BYTES   49152           // + 8 waves x 1KB P staging

typedef __attribute__((ext_vector_type(16))) __bf16 v16bf;
typedef __attribute__((ext_vector_type(8)))  float  v8f;

__device__ __forceinline__ __bf16 f2bf(float f) {
    unsigned u = __builtin_bit_cast(unsigned, f);
    unsigned r = u + 0x7FFFu + ((u >> 16) & 1u);   // round-to-nearest-even
    return __builtin_bit_cast(__bf16, (unsigned short)(r >> 16));
}
__device__ __forceinline__ float bf2f(__bf16 x) {
    unsigned u = ((unsigned)__builtin_bit_cast(unsigned short, x)) << 16;
    return __builtin_bit_cast(float, u);
}
__device__ __forceinline__ v8f v8f_zero() {
    v8f v;
#pragma unroll
    for (int i = 0; i < 8; ++i) v[i] = 0.0f;
    return v;
}

// CDNA5 async global->LDS copy (ASYNCcnt path), 16B per lane.
__device__ __forceinline__ void async_ld_b128(unsigned lds_off, const void* gsrc) {
    asm volatile("global_load_async_to_lds_b128 %0, %1, off"
                 :: "v"(lds_off), "v"(gsrc)
                 : "memory");
}
__device__ __forceinline__ void wait_async0() {
    asm volatile("s_wait_asynccnt 0x0" ::: "memory");
}

// ---------------------------------------------------------------------------
// One-time fp32 -> bf16 conversion (vectorized) and fp32 -> bf16 transpose
// ([K,N] row-major -> [N,K] K-major) so GEMM B-fragments are contiguous.
// ---------------------------------------------------------------------------
__global__ __launch_bounds__(256) void cvt_f32_to_bf16(
    const float* __restrict__ in, __bf16* __restrict__ out, int n) {
    int i = (blockIdx.x * blockDim.x + threadIdx.x) * 4;
    if (i + 3 < n) {
        float4 v = *(const float4*)(in + i);
        out[i + 0] = f2bf(v.x);
        out[i + 1] = f2bf(v.y);
        out[i + 2] = f2bf(v.z);
        out[i + 3] = f2bf(v.w);
    } else {
        for (; i < n; ++i) out[i] = f2bf(in[i]);
    }
}

__global__ __launch_bounds__(256) void cvt_transpose_bf16(
    const float* __restrict__ in, __bf16* __restrict__ out, int K, int N) {
    int idx = blockIdx.x * blockDim.x + threadIdx.x;
    if (idx >= K * N) return;
    int k = idx / N, n = idx - k * N;
    out[(size_t)n * K + k] = f2bf(in[idx]);
}

// ---------------------------------------------------------------------------
// bf16-WMMA GEMM: C[M,N] = A[M,K] @ Bt[N,K]^T (+bias). Both operands bf16,
// B pre-transposed so every fragment is a contiguous 32B per-lane load.
// Wave computes 16(M) x 64(N); block = 8 waves = 128(M).
// OUT_MODE: 0 = fp32, 1 = fp32 + bias, 2 = bf16 row-major,
//           3 = bf16 transposed-with-batch [B][N][T] (for V).
// A/B fragments: contiguous k = 16*half + e in both; C/D layout per CDNA5 ISA 7.12.2 (wave32).
// ---------------------------------------------------------------------------
enum { OUT_F32 = 0, OUT_F32_BIAS = 1, OUT_BF16 = 2, OUT_BF16_TBH = 3 };

template <int OUT_MODE>
__global__ __launch_bounds__(256) void gemm_bf16_wmma(
    const __bf16* __restrict__ A, const __bf16* __restrict__ Bt,
    float* __restrict__ outF, __bf16* __restrict__ outB,
    const float* __restrict__ bias, int M, int N, int K) {
    const int lane = threadIdx.x & 31;
    const int wave = threadIdx.x >> 5;
    const int l15  = lane & 15;
    const int hlf  = lane >> 4;
    const int m0 = blockIdx.y * 128 + wave * 16;
    const int n0 = blockIdx.x * 64;
    if (m0 >= M || n0 >= N) return;

    v8f acc[4];
#pragma unroll
    for (int nt = 0; nt < 4; ++nt) acc[nt] = v8f_zero();

    const __bf16* arow  = A + (size_t)(m0 + l15) * K + hlf * 16;   // A k-base
    const __bf16* brow0 = Bt + (size_t)(n0 + l15) * K + hlf * 16;  // B k-base
    const size_t bstep = (size_t)16 * K;

    for (int k = 0; k < K; k += 32) {
        __builtin_prefetch(arow + k + 256, 0, 1);
        __builtin_prefetch(brow0 + k + 256, 0, 1);
        v16bf af;
#pragma unroll
        for (int e = 0; e < 8; ++e) {
            af[e]     = arow[k + e];
            af[8 + e] = arow[k + 8 + e];
        }
#pragma unroll
        for (int nt = 0; nt < 4; ++nt) {
            const __bf16* bp = brow0 + (size_t)nt * bstep + k;
            v16bf bfr;
#pragma unroll
            for (int e = 0; e < 16; ++e) bfr[e] = bp[e];
            acc[nt] = __builtin_amdgcn_wmma_f32_16x16x32_bf16(
                false, af, false, bfr, (short)0, acc[nt], false, false);
        }
    }
#pragma unroll
    for (int nt = 0; nt < 4; ++nt) {
#pragma unroll
        for (int r = 0; r < 8; ++r) {
            const size_t row = (size_t)m0 + hlf * 8 + r;
            const int col = n0 + nt * 16 + l15;
            float v = acc[nt][r];
            if constexpr (OUT_MODE == OUT_F32_BIAS) v += bias[col];
            if constexpr (OUT_MODE == OUT_F32 || OUT_MODE == OUT_F32_BIAS) {
                outF[row * (size_t)N + col] = v;
            } else if constexpr (OUT_MODE == OUT_BF16) {
                outB[row * (size_t)N + col] = f2bf(v);
            } else {  // OUT_BF16_TBH: [B][N][T]; contiguous in t per lane
                const int bb = (int)(row >> 11);      // row / TT
                const int t  = (int)(row & (TT - 1)); // row % TT
                outB[((size_t)bb * N + col) * TT + t] = f2bf(v);
            }
        }
    }
}

// ---------------------------------------------------------------------------
// RoPE (interleaved rotate, positions along T) for k_R [M,64] and
// q_R [M, H*64]; bf16 in, bf16 out (fp32 math).
// ---------------------------------------------------------------------------
__global__ __launch_bounds__(256) void rope_pack_kernel(
    const __bf16* __restrict__ kRp, const __bf16* __restrict__ qRp,
    __bf16* __restrict__ kRb, __bf16* __restrict__ QRb) {
    const int bt = blockIdx.x;
    const float tf = (float)(bt % TT);
    const float klog = 9.210340371976184f / 32.0f;   // ln(10000)/half

    for (int i = threadIdx.x; i < 32; i += blockDim.x) {
        float inv = __expf(-(float)i * klog);
        float sn, cs;
        __sincosf(tf * inv, &sn, &cs);
        size_t base = (size_t)bt * DRn;
        float x1 = bf2f(kRp[base + i]);
        float x2 = bf2f(kRp[base + 32 + i]);
        kRb[base + 2 * i]     = f2bf(x1 * cs - x2 * sn);
        kRb[base + 2 * i + 1] = f2bf(x1 * sn + x2 * cs);
    }
    for (int j = threadIdx.x; j < HHn * 32; j += blockDim.x) {
        int hh = j >> 5, i = j & 31;
        float inv = __expf(-(float)i * klog);
        float sn, cs;
        __sincosf(tf * inv, &sn, &cs);
        size_t base = (size_t)bt * (HHn * DRn) + (size_t)hh * DRn;
        float x1 = bf2f(qRp[base + i]);
        float x2 = bf2f(qRp[base + 32 + i]);
        QRb[base + 2 * i]     = f2bf(x1 * cs - x2 * sn);
        QRb[base + 2 * i + 1] = f2bf(x1 * sn + x2 * cs);
    }
}

// ---------------------------------------------------------------------------
// Fused flash attention. Block = 8 waves, ALL on the same (b,h); each wave
// owns one 16-query tile. Per 32-key chunk the block stages the shared
// Kf tile (32 x 192, Kc||kR) and V tile (128 x 32, key-contiguous from the
// pre-transposed Vt) into LDS ONCE via async global->LDS copies
// (double-buffered, ASYNCcnt + barrier), killing the 8x redundant global
// K/V traffic. Scores via 12 WMMAs/chunk, online softmax, P through LDS,
// P@V via 8 WMMAs. ctx accumulated fp32, emitted bf16.
// ---------------------------------------------------------------------------
__global__ __launch_bounds__(256) void mla_flash_kernel(
    const __bf16* __restrict__ Qc, const __bf16* __restrict__ QR,
    const __bf16* __restrict__ Kc, const __bf16* __restrict__ kR,
    const __bf16* __restrict__ Vt, __bf16* __restrict__ ctx) {
    extern __shared__ char smem[];
    const int tid  = threadIdx.x;
    const int lane = tid & 31;
    const int wave = tid >> 5;
    const int l15  = lane & 15;
    const int hlf  = lane >> 4;
    const int h = blockIdx.y;
    const int b = blockIdx.z;
    const int q0 = (blockIdx.x * 8 + wave) * 16;
    const float scale = 0.07216878364870323f;  // 1/sqrt(192)

    // ---- async-stage one 32-key chunk (K: 768 x b128, V: 512 x b128) ----
    auto issue_tile = [&](int kc, int p) {
#pragma unroll
        for (int t = 0; t < 3; ++t) {   // K tile rows [key][192 d]
            int i = tid + t * 256;
            int key = i / 24, seg = i - key * 24;   // 24 x 16B per row
            unsigned lofs = (unsigned)(p * K_TILE_BYTES + key * 384 + seg * 16);
            const void* src;
            if (seg < 16)
                src = (const void*)(Kc + ((size_t)b * TT + kc + key) * DDm +
                                    h * HDn + seg * 8);
            else
                src = (const void*)(kR + ((size_t)b * TT + kc + key) * DRn +
                                    (seg - 16) * 8);
            async_ld_b128(lofs, src);
        }
#pragma unroll
        for (int t = 0; t < 2; ++t) {   // V tile rows [d][32 keys]
            int i = tid + t * 256;
            int d = i >> 2, seg = i & 3;            // 4 x 16B per row
            unsigned lofs = (unsigned)(V_TILE_OFF + p * V_TILE_BYTES + d * 64 + seg * 16);
            const void* src =
                (const void*)(Vt + ((size_t)b * DDm + h * HDn + d) * TT + kc + seg * 8);
            async_ld_b128(lofs, src);
        }
    };

    // ---- preload Q fragments (scale folded in) ----
    v16bf qf[6];
    {
        const size_t row = (size_t)b * TT + q0 + l15;
        const __bf16* qcp = Qc + row * (size_t)DDm + h * HDn;
        const __bf16* qrp = QR + row * (size_t)(HHn * DRn) + h * DRn;
        const int kb = hlf * 16;
#pragma unroll
        for (int ks = 0; ks < 6; ++ks) {
            const __bf16* src = (ks < 4) ? (qcp + ks * 32 + kb)
                                         : (qrp + (ks - 4) * 32 + kb);
#pragma unroll
            for (int e = 0; e < 8; ++e) {
                qf[ks][e]     = f2bf(bf2f(src[e])      * scale);
                qf[ks][8 + e] = f2bf(bf2f(src[8 + e]) * scale);
            }
        }
    }

    v8f acc[8];
#pragma unroll
    for (int j = 0; j < 8; ++j) acc[j] = v8f_zero();
    float mrow[8], lrow[8];
#pragma unroll
    for (int r = 0; r < 8; ++r) { mrow[r] = -1e30f; lrow[r] = 0.0f; }

    const int bkb = hlf * 16;
    __bf16* Pw = (__bf16*)(smem + P_OFF + wave * 1024);   // [16][32]

    issue_tile(0, 0);   // prologue

    for (int ic = 0; ic < TT / 32; ++ic) {
        const int p = ic & 1;
        wait_async0();        // my tile writes for buffer p are done
        __syncthreads();      // everyone's writes done; prev reader of p^1 done
        if (ic + 1 < TT / 32) issue_tile((ic + 1) * 32, p ^ 1);

        const __bf16* Ks = (const __bf16*)(smem + p * K_TILE_BYTES);
        const __bf16* Vs = (const __bf16*)(smem + V_TILE_OFF + p * V_TILE_BYTES);

        // ---- scores S[16 x 32] = Qf @ Kf^T  (Kf from LDS) ----
        v8f s0 = v8f_zero(), s1 = v8f_zero();
        const __bf16* kr0 = Ks + (size_t)l15 * 192 + bkb;
        const __bf16* kr1 = Ks + (size_t)(16 + l15) * 192 + bkb;
#pragma unroll
        for (int ks = 0; ks < 6; ++ks) {
            v16bf kf0, kf1;
#pragma unroll
            for (int e = 0; e < 16; ++e) {
                kf0[e] = kr0[ks * 32 + e];
                kf1[e] = kr1[ks * 32 + e];
            }
            s0 = __builtin_amdgcn_wmma_f32_16x16x32_bf16(
                false, qf[ks], false, kf0, (short)0, s0, false, false);
            s1 = __builtin_amdgcn_wmma_f32_16x16x32_bf16(
                false, qf[ks], false, kf1, (short)0, s1, false, false);
        }
        // ---- online softmax (rows live on 16-lane halves) ----
#pragma unroll
        for (int r = 0; r < 8; ++r) {
            float mx = fmaxf(s0[r], s1[r]);
            mx = fmaxf(mx, __shfl_xor(mx, 1));
            mx = fmaxf(mx, __shfl_xor(mx, 2));
            mx = fmaxf(mx, __shfl_xor(mx, 4));
            mx = fmaxf(mx, __shfl_xor(mx, 8));
            float mn  = fmaxf(mrow[r], mx);
            float al  = __expf(mrow[r] - mn);
            float p0e = __expf(s0[r] - mn);
            float p1e = __expf(s1[r] - mn);
            float sm  = p0e + p1e;
            sm += __shfl_xor(sm, 1);
            sm += __shfl_xor(sm, 2);
            sm += __shfl_xor(sm, 4);
            sm += __shfl_xor(sm, 8);
            lrow[r] = lrow[r] * al + sm;
            mrow[r] = mn;
            const int m = r + hlf * 8;
            Pw[m * 32 + l15]      = f2bf(p0e);
            Pw[m * 32 + 16 + l15] = f2bf(p1e);
#pragma unroll
            for (int j = 0; j < 8; ++j) acc[j][r] *= al;
        }
        // per-wave LDS ops are in-order; only block compiler reordering.
        asm volatile("" ::: "memory");
        v16bf pf;
        {
            const __bf16* pp = Pw + l15 * 32 + hlf * 16;
#pragma unroll
            for (int e = 0; e < 8; ++e) { pf[e] = pp[e]; pf[8 + e] = pp[8 + e]; }
        }
        asm volatile("" ::: "memory");
        // ---- ctx += P @ V  (V fragment from LDS, key-contiguous) ----
#pragma unroll
        for (int j = 0; j < 8; ++j) {
            const __bf16* vp = Vs + (size_t)(j * 16 + l15) * 32 + bkb;
            v16bf vf;
#pragma unroll
            for (int e = 0; e < 16; ++e) vf[e] = vp[e];
            acc[j] = __builtin_amdgcn_wmma_f32_16x16x32_bf16(
                false, pf, false, vf, (short)0, acc[j], false, false);
        }
    }
    // ---- normalize and store ctx as [B,T,H,HD] (== [M, D]) bf16 ----
#pragma unroll
    for (int r = 0; r < 8; ++r) {
        const float inv = 1.0f / lrow[r];
        const size_t row = (size_t)b * TT + q0 + hlf * 8 + r;
        __bf16* op = ctx + row * (size_t)DDm + h * HDn + l15;
#pragma unroll
        for (int j = 0; j < 8; ++j) op[j * 16] = f2bf(acc[j][r] * inv);
    }
}

// ---------------------------------------------------------------------------
extern "C" void kernel_launch(void* const* d_in, const int* in_sizes, int n_in,
                              void* d_out, int out_size, void* d_ws, size_t ws_size,
                              hipStream_t stream) {
    (void)in_sizes; (void)n_in; (void)out_size; (void)ws_size;
    const float* x     = (const float*)d_in[0];
    const float* W_DKV = (const float*)d_in[1];
    const float* W_DQ  = (const float*)d_in[2];
    const float* W_UK  = (const float*)d_in[3];
    const float* W_UV  = (const float*)d_in[4];
    const float* W_UQ  = (const float*)d_in[5];
    const float* W_KR  = (const float*)d_in[6];
    const float* W_QR  = (const float*)d_in[7];
    const float* W_O   = (const float*)d_in[8];
    const float* b_O   = (const float*)d_in[9];
    float* out = (float*)d_out;

    char* p = (char*)d_ws;
    auto take = [&](size_t bytes) {
        char* q = p;
        p += (bytes + 255) & ~(size_t)255;
        return q;
    };
    __bf16* xb    = (__bf16*)take((size_t)MMr * DDm * 2);
    __bf16* WDKVt = (__bf16*)take((size_t)DDm * DCn * 2);
    __bf16* WDQt  = (__bf16*)take((size_t)DDm * DCn * 2);
    __bf16* WKRt  = (__bf16*)take((size_t)DDm * DRn * 2);
    __bf16* WUKt  = (__bf16*)take((size_t)DCn * DDm * 2);
    __bf16* WUVt  = (__bf16*)take((size_t)DCn * DDm * 2);
    __bf16* WUQt  = (__bf16*)take((size_t)DCn * DDm * 2);
    __bf16* WQRt  = (__bf16*)take((size_t)DCn * HHn * DRn * 2);
    __bf16* WOt   = (__bf16*)take((size_t)DDm * DDm * 2);
    __bf16* c_kv  = (__bf16*)take((size_t)MMr * DCn * 2);
    __bf16* c_q   = (__bf16*)take((size_t)MMr * DCn * 2);
    __bf16* kRp   = (__bf16*)take((size_t)MMr * DRn * 2);
    __bf16* qRp   = (__bf16*)take((size_t)MMr * HHn * DRn * 2);
    __bf16* Kc    = (__bf16*)take((size_t)MMr * DDm * 2);
    __bf16* Vt    = (__bf16*)take((size_t)MMr * DDm * 2);
    __bf16* Qc    = (__bf16*)take((size_t)MMr * DDm * 2);
    __bf16* QR    = (__bf16*)take((size_t)MMr * HHn * DRn * 2);
    __bf16* kRb   = (__bf16*)take((size_t)MMr * DRn * 2);
    __bf16* ctxb  = (__bf16*)take((size_t)MMr * DDm * 2);

    dim3 blk(256);
    // one-time bf16 conversions (x plain; weights transposed to [N,K])
    {
        int n = MMr * DDm;
        cvt_f32_to_bf16<<<dim3((n / 4 + 255) / 256), blk, 0, stream>>>(x, xb, n);
        auto tr = [&](const float* src, __bf16* dst, int K, int N) {
            cvt_transpose_bf16<<<dim3((K * N + 255) / 256), blk, 0, stream>>>(src, dst, K, N);
        };
        tr(W_DKV, WDKVt, DDm, DCn);
        tr(W_DQ,  WDQt,  DDm, DCn);
        tr(W_KR,  WKRt,  DDm, DRn);
        tr(W_UK,  WUKt,  DCn, DDm);
        tr(W_UV,  WUVt,  DCn, DDm);
        tr(W_UQ,  WUQt,  DCn, DDm);
        tr(W_QR,  WQRt,  DCn, HHn * DRn);
        tr(W_O,   WOt,   DDm, DDm);
    }
    // down projections (K = D = 2048) -> bf16
    gemm_bf16_wmma<OUT_BF16><<<dim3(DCn / 64, MMr / 128), blk, 0, stream>>>(
        xb, WDKVt, nullptr, c_kv, nullptr, MMr, DCn, DDm);
    gemm_bf16_wmma<OUT_BF16><<<dim3(DCn / 64, MMr / 128), blk, 0, stream>>>(
        xb, WDQt, nullptr, c_q, nullptr, MMr, DCn, DDm);
    gemm_bf16_wmma<OUT_BF16><<<dim3(DRn / 64, MMr / 128), blk, 0, stream>>>(
        xb, WKRt, nullptr, kRp, nullptr, MMr, DRn, DDm);
    // up projections (K = DC = 512) -> bf16 (V transposed to [B][D][T])
    gemm_bf16_wmma<OUT_BF16><<<dim3(DDm / 64, MMr / 128), blk, 0, stream>>>(
        c_kv, WUKt, nullptr, Kc, nullptr, MMr, DDm, DCn);
    gemm_bf16_wmma<OUT_BF16_TBH><<<dim3(DDm / 64, MMr / 128), blk, 0, stream>>>(
        c_kv, WUVt, nullptr, Vt, nullptr, MMr, DDm, DCn);
    gemm_bf16_wmma<OUT_BF16><<<dim3(DDm / 64, MMr / 128), blk, 0, stream>>>(
        c_q, WUQt, nullptr, Qc, nullptr, MMr, DDm, DCn);
    gemm_bf16_wmma<OUT_BF16><<<dim3((HHn * DRn) / 64, MMr / 128), blk, 0, stream>>>(
        c_q, WQRt, nullptr, qRp, nullptr, MMr, HHn * DRn, DCn);
    // RoPE + pack
    rope_pack_kernel<<<dim3(MMr), blk, 0, stream>>>(kRp, qRp, kRb, QR);
    // fused flash attention (async double-buffered K/V tiles in LDS)
    mla_flash_kernel<<<dim3(TT / 128, HHn, BB), blk, SMEM_BYTES, stream>>>(
        Qc, QR, Kc, kRb, Vt, ctxb);
    // output projection with bias -> d_out fp32
    gemm_bf16_wmma<OUT_F32_BIAS><<<dim3(DDm / 64, MMr / 128), blk, 0, stream>>>(
        ctxb, WOt, out, nullptr, b_O, MMr, DDm, DDm);
}
